// RoleActorTrans_1108101562455
// MI455X (gfx1250) — matrix operand order repaired and checked by reading, not compiled, not run on hardware
//
#include <hip/hip_runtime.h>
#include <hip/hip_bf16.h>

// ---------------------------------------------------------------------------
// Role-actor forward for MI455X (gfx1250): bf16 WMMA everywhere.
// T=32, B=1024 -> TB=32768 tokens; D_OBS=256, H=512, A=32, N_ROLES=6.
// ---------------------------------------------------------------------------

#define TB      32768
#define DOBS    256
#define H       512
#define ACT     32
#define NROLES  6
#define TILES   (TB / 16)   // 2048 16-row M tiles

typedef __bf16 v16bf __attribute__((ext_vector_type(16)));
typedef float  v8f   __attribute__((ext_vector_type(8)));

union Frag { v16bf v; uint4 q[2]; };

static __device__ __forceinline__ unsigned short f2bf(float f) {
    union { float f; unsigned u; } v; v.f = f;
    unsigned r = v.u + 0x7FFFu + ((v.u >> 16) & 1u);
    return (unsigned short)(r >> 16);
}
static __device__ __forceinline__ float bf2f(unsigned short h) {
    union { unsigned u; float f; } v; v.u = ((unsigned)h) << 16; return v.f;
}

static __device__ __forceinline__ v8f wmma_bf16(const Frag& a, const Frag& b, v8f c) {
    return __builtin_amdgcn_wmma_f32_16x16x32_bf16(false, a.v, false, b.v,
                                                   (short)0, c, false, false);
}

// A fragment (16x32 bf16, M x K): lane<16 -> row=lane, K = {kk..kk+7, kk+16..kk+23}
//                                 lane>=16 -> same row, K = {kk+8..kk+15, kk+24..kk+31}
static __device__ __forceinline__ Frag load_a_global(const unsigned short* __restrict__ A,
                                                     int row, int K, int kk, int lane) {
    Frag f;
    const int ksel = (lane >> 4) * 8;
    const unsigned short* base = A + (size_t)row * K + kk + ksel;
    f.q[0] = *reinterpret_cast<const uint4*>(base);
    f.q[1] = *reinterpret_cast<const uint4*>(base + 16);
    return f;
}

// B fragment (32x16 bf16, K x N) from W^T (N x K row-major):
// lane<16 -> col=lane, K = {kk..kk+7, kk+16..kk+23} ; lane>=16 -> col=lane-16, K = {kk+8..kk+15, kk+24..kk+31}
static __device__ __forceinline__ Frag load_b_wt(const unsigned short* __restrict__ WT,
                                                 int col, int K, int kk, int lane) {
    Frag f;
    const int kb = (lane >> 4) * 8;
    const unsigned short* base = WT + (size_t)col * K + kk + kb;
    f.q[0] = *reinterpret_cast<const uint4*>(base);
    f.q[1] = *reinterpret_cast<const uint4*>(base + 16);
    return f;
}

static __device__ __forceinline__ v8f vzero8() {
    v8f z = {0.f, 0.f, 0.f, 0.f, 0.f, 0.f, 0.f, 0.f};
    return z;
}

// ---------------------------------------------------------------------------
// Weight convert+transpose: src (batch,K,N) f32 -> dst (batch,N,K) bf16
// ---------------------------------------------------------------------------
__global__ void transpose_bf16_kernel(const float* __restrict__ src,
                                      unsigned short* __restrict__ dst,
                                      int K, int N) {
    const int b = blockIdx.y;
    const size_t kn = (size_t)K * N;
    size_t idx = (size_t)blockIdx.x * 256 + threadIdx.x;
    if (idx >= kn) return;
    int k = (int)(idx / N);
    int n = (int)(idx % N);
    dst[b * kn + (size_t)n * K + k] = f2bf(src[b * kn + idx]);
}

// ---------------------------------------------------------------------------
// LayerNorm over obs (256) -> bf16. One wave per token, 8 tokens per block.
// ---------------------------------------------------------------------------
__global__ __launch_bounds__(256) void ln_obs_kernel(const float* __restrict__ obs,
                                                     const float* __restrict__ fs,
                                                     const float* __restrict__ fb,
                                                     unsigned short* __restrict__ out) {
    const int lane = threadIdx.x & 31;
    const int w    = threadIdx.x >> 5;
    const int tok  = blockIdx.x * 8 + w;
    const float* row = obs + (size_t)tok * DOBS;

    float4 q0 = *reinterpret_cast<const float4*>(row + lane * 8);
    float4 q1 = *reinterpret_cast<const float4*>(row + lane * 8 + 4);
    float x[8] = {q0.x, q0.y, q0.z, q0.w, q1.x, q1.y, q1.z, q1.w};

    float s = 0.f, sq = 0.f;
#pragma unroll
    for (int i = 0; i < 8; ++i) { s += x[i]; sq += x[i] * x[i]; }
#pragma unroll
    for (int m = 16; m >= 1; m >>= 1) {
        s  += __shfl_xor(s,  m, 32);
        sq += __shfl_xor(sq, m, 32);
    }
    const float mean = s * (1.f / DOBS);
    const float var  = sq * (1.f / DOBS) - mean * mean;
    const float rs   = rsqrtf(var + 1e-5f);

    union { unsigned short h[8]; uint4 q; } o;
#pragma unroll
    for (int i = 0; i < 8; ++i) {
        int c = lane * 8 + i;
        o.h[i] = f2bf((x[i] - mean) * rs * fs[c] + fb[c]);
    }
    *reinterpret_cast<uint4*>(out + (size_t)tok * DOBS + lane * 8) = o.q;
}

// ---------------------------------------------------------------------------
// Fused GEMM (M-tile=16, N=512) + bias + ReLU + LayerNorm(512) -> bf16.
// 8 waves/block; wave w owns columns [64w, 64w+64).
// ---------------------------------------------------------------------------
__global__ __launch_bounds__(256) void gemm_relu_ln_512(
    const unsigned short* __restrict__ A,   // (TB, K) bf16
    const unsigned short* __restrict__ WT,  // (512, K) bf16
    const float* __restrict__ bias,
    const float* __restrict__ gamma,
    const float* __restrict__ beta,
    unsigned short* __restrict__ out,       // (TB, 512) bf16
    int K) {
    __shared__ float tile[16 * 512];

    const int tid  = threadIdx.x;
    const int lane = tid & 31;
    const int w    = tid >> 5;
    const int hi   = lane >> 4;
    const int l15  = lane & 15;
    const int m0   = blockIdx.x * 16;

    v8f acc[4] = {vzero8(), vzero8(), vzero8(), vzero8()};

    for (int kk = 0; kk < K; kk += 32) {
        Frag a = load_a_global(A, m0 + l15, K, kk, lane);
#pragma unroll
        for (int nt = 0; nt < 4; ++nt) {
            Frag b = load_b_wt(WT, w * 64 + nt * 16 + l15, K, kk, lane);
            acc[nt] = wmma_bf16(a, b, acc[nt]);
        }
    }

#pragma unroll
    for (int nt = 0; nt < 4; ++nt) {
        const int col = w * 64 + nt * 16 + l15;
        const float bi = bias[col];
#pragma unroll
        for (int j = 0; j < 8; ++j) {
            int r = j + hi * 8;
            tile[r * 512 + col] = fmaxf(acc[nt][j] + bi, 0.f);
        }
    }
    __syncthreads();

    // LayerNorm: wave w handles rows 2w, 2w+1; lane covers cols [16*lane, 16*lane+16)
#pragma unroll
    for (int rr = 0; rr < 2; ++rr) {
        const int r = w * 2 + rr;
        float x[16], s = 0.f, sq = 0.f;
#pragma unroll
        for (int i = 0; i < 16; ++i) {
            x[i] = tile[r * 512 + lane * 16 + i];
            s += x[i]; sq += x[i] * x[i];
        }
#pragma unroll
        for (int m = 16; m >= 1; m >>= 1) {
            s  += __shfl_xor(s,  m, 32);
            sq += __shfl_xor(sq, m, 32);
        }
        const float mean = s * (1.f / 512.f);
        const float var  = sq * (1.f / 512.f) - mean * mean;
        const float rs   = rsqrtf(var + 1e-5f);

        union { unsigned short h[16]; uint4 q[2]; } o;
#pragma unroll
        for (int i = 0; i < 16; ++i) {
            int c = lane * 16 + i;
            o.h[i] = f2bf((x[i] - mean) * rs * gamma[c] + beta[c]);
        }
        uint4* dst = reinterpret_cast<uint4*>(out + (size_t)(m0 + r) * 512 + lane * 16);
        dst[0] = o.q[0];
        dst[1] = o.q[1];
    }
}

// ---------------------------------------------------------------------------
// Per-role route, fused: r = relu(obs_n @ Wr0[k] + br0) ; r2 = relu(r @ Wr1[k] + br1)
// then e_final[t] = e1[t] + r2[t] stored only where role_ids[t]==k.
// Grid: (TILES, NROLES). Phase A: each wave 1 16-col tile of r (K=256).
// Phase B: r lives in LDS (bf16), each wave 4 16-col tiles of r2 (K=128).
// ---------------------------------------------------------------------------
__global__ __launch_bounds__(256) void route_kernel(
    const unsigned short* __restrict__ obs_n,  // (TB,256) bf16
    const unsigned short* __restrict__ Wr0T,   // (6,128,256) bf16
    const float* __restrict__ br0,             // (6,128)
    const unsigned short* __restrict__ Wr1T,   // (6,512,128) bf16
    const float* __restrict__ br1,             // (6,512)
    const unsigned short* __restrict__ e1,     // (TB,512) bf16
    const int* __restrict__ role_ids,          // (TB)
    unsigned short* __restrict__ ef) {         // (TB,512) bf16
    __shared__ unsigned short rsh[16 * 128];
    __shared__ int roles[16];

    const int tid  = threadIdx.x;
    const int lane = tid & 31;
    const int w    = tid >> 5;
    const int hi   = lane >> 4;
    const int l15  = lane & 15;
    const int k    = blockIdx.y;
    const int m0   = blockIdx.x * 16;

    if (tid < 16) roles[tid] = role_ids[m0 + tid];

    // ---- Phase A: r tile (16 x 128), wave w -> cols [16w, 16w+16)
    v8f acc = vzero8();
    const unsigned short* W0k = Wr0T + (size_t)k * 128 * 256;
    for (int kk = 0; kk < 256; kk += 32) {
        Frag a = load_a_global(obs_n, m0 + l15, 256, kk, lane);
        Frag b = load_b_wt(W0k, w * 16 + l15, 256, kk, lane);
        acc = wmma_bf16(a, b, acc);
    }
    {
        const int col = w * 16 + l15;
        const float bi = br0[k * 128 + col];
#pragma unroll
        for (int j = 0; j < 8; ++j) {
            int r = j + hi * 8;
            rsh[r * 128 + col] = f2bf(fmaxf(acc[j] + bi, 0.f));
        }
    }
    __syncthreads();

    // ---- Phase B: r2 tile (16 x 512), A from LDS
    v8f acc2[4] = {vzero8(), vzero8(), vzero8(), vzero8()};
    const unsigned short* W1k = Wr1T + (size_t)k * 512 * 128;
    for (int kk = 0; kk < 128; kk += 32) {
        Frag a;
        const int ksel = hi * 8;
        a.q[0] = *reinterpret_cast<const uint4*>(&rsh[l15 * 128 + kk + ksel]);
        a.q[1] = *reinterpret_cast<const uint4*>(&rsh[l15 * 128 + kk + 16 + ksel]);
#pragma unroll
        for (int nt = 0; nt < 4; ++nt) {
            Frag b = load_b_wt(W1k, w * 64 + nt * 16 + l15, 128, kk, lane);
            acc2[nt] = wmma_bf16(a, b, acc2[nt]);
        }
    }

#pragma unroll
    for (int nt = 0; nt < 4; ++nt) {
        const int col = w * 64 + nt * 16 + l15;
        const float bi = br1[k * 512 + col];
#pragma unroll
        for (int j = 0; j < 8; ++j) {
            int r = j + hi * 8;
            if (roles[r] == k) {
                size_t off = (size_t)(m0 + r) * 512 + col;
                float v = fmaxf(acc2[nt][j] + bi, 0.f) + bf2f(e1[off]);
                ef[off] = f2bf(v);
            }
        }
    }
}

// ---------------------------------------------------------------------------
// Per-role head: h0 = relu(ef @ Wh0[k]) via WMMA (K=512, N=64), then scalar
// 64->32->32, role-gather + availability mask -> logits (f32).
// Grid: (TILES, NROLES), 1 wave per block.
// ---------------------------------------------------------------------------
__global__ __launch_bounds__(32) void head_kernel(
    const unsigned short* __restrict__ ef,     // (TB,512) bf16
    const unsigned short* __restrict__ Wh0T,   // (6,64,512) bf16
    const float* __restrict__ bh0,             // (6,64)
    const float* __restrict__ Wh1,             // (6,64,32)
    const float* __restrict__ bh1,             // (6,32)
    const float* __restrict__ Wh2,             // (6,32,32)
    const float* __restrict__ bh2,             // (6,32)
    const int* __restrict__ role_ids,          // (TB)
    const float* __restrict__ avail,           // (TB,32)
    float* __restrict__ logits) {              // (TB,32)
    __shared__ float h0s[16 * 64];
    __shared__ float h1s[16 * 32];
    __shared__ int roles[16];

    const int lane = threadIdx.x;
    const int hi   = lane >> 4;
    const int l15  = lane & 15;
    const int k    = blockIdx.y;
    const int m0   = blockIdx.x * 16;

    if (lane < 16) roles[lane] = role_ids[m0 + lane];

    v8f acc[4] = {vzero8(), vzero8(), vzero8(), vzero8()};
    const unsigned short* Wk = Wh0T + (size_t)k * 64 * 512;
    for (int kk = 0; kk < 512; kk += 32) {
        Frag a = load_a_global(ef, m0 + l15, 512, kk, lane);
#pragma unroll
        for (int nt = 0; nt < 4; ++nt) {
            Frag b = load_b_wt(Wk, nt * 16 + l15, 512, kk, lane);
            acc[nt] = wmma_bf16(a, b, acc[nt]);
        }
    }
#pragma unroll
    for (int nt = 0; nt < 4; ++nt) {
        const int col = nt * 16 + l15;
        const float bi = bh0[k * 64 + col];
#pragma unroll
        for (int j = 0; j < 8; ++j) {
            int r = j + hi * 8;
            h0s[r * 64 + col] = fmaxf(acc[nt][j] + bi, 0.f);
        }
    }
    __syncthreads();

    // h1 = relu(h0 @ Wh1[k] + bh1[k]) : lane = output channel (32)
    const float* W1k = Wh1 + (size_t)k * 64 * 32;
    for (int t = 0; t < 16; ++t) {
        float s = bh1[k * 32 + lane];
#pragma unroll 8
        for (int f = 0; f < 64; ++f) s += h0s[t * 64 + f] * W1k[f * 32 + lane];
        h1s[t * 32 + lane] = fmaxf(s, 0.f);
    }
    __syncthreads();

    // logits = h1 @ Wh2[k] + bh2[k], gather + mask
    const float* W2k = Wh2 + (size_t)k * 32 * 32;
    for (int t = 0; t < 16; ++t) {
        if (roles[t] == k) {
            float s = bh2[k * 32 + lane];
#pragma unroll 8
            for (int g = 0; g < 32; ++g) s += h1s[t * 32 + g] * W2k[g * 32 + lane];
            size_t off = (size_t)(m0 + t) * ACT + lane;
            logits[off] = (avail[off] > 0.5f) ? s : -1e10f;
        }
    }
}

// ---------------------------------------------------------------------------
// Host launcher
// ---------------------------------------------------------------------------
extern "C" void kernel_launch(void* const* d_in, const int* in_sizes, int n_in,
                              void* d_out, int out_size, void* d_ws, size_t ws_size,
                              hipStream_t stream) {
    const float* rnn       = (const float*)d_in[0];
    const float* obs       = (const float*)d_in[1];
    const float* avail     = (const float*)d_in[3];
    const int*   role_ids  = (const int*)  d_in[4];
    const float* fn_scale  = (const float*)d_in[5];
    const float* fn_bias   = (const float*)d_in[6];
    const float* W0        = (const float*)d_in[7];
    const float* b0        = (const float*)d_in[8];
    const float* ln0_s     = (const float*)d_in[9];
    const float* ln0_b     = (const float*)d_in[10];
    const float* W1        = (const float*)d_in[11];
    const float* b1        = (const float*)d_in[12];
    const float* ln1_s     = (const float*)d_in[13];
    const float* ln1_b     = (const float*)d_in[14];
    const float* Wr0       = (const float*)d_in[15];
    const float* br0       = (const float*)d_in[16];
    const float* Wr1       = (const float*)d_in[17];
    const float* br1       = (const float*)d_in[18];
    const float* Wh0       = (const float*)d_in[19];
    const float* bh0       = (const float*)d_in[20];
    const float* Wh1       = (const float*)d_in[21];
    const float* bh1       = (const float*)d_in[22];
    const float* Wh2       = (const float*)d_in[23];
    const float* bh2       = (const float*)d_in[24];

    // workspace layout (bf16 stored as ushort)
    char* ws = (char*)d_ws;
    unsigned short* obs_n = (unsigned short*)(ws);                          // TB*256
    unsigned short* e0    = (unsigned short*)(ws + 16777216);               // TB*512 (reused as e_final)
    unsigned short* e1    = (unsigned short*)(ws + 16777216 + 33554432);    // TB*512
    char* wbase = ws + 16777216 + 2u * 33554432u;
    unsigned short* W0T  = (unsigned short*)(wbase);                        // 512x256
    unsigned short* W1T  = (unsigned short*)(wbase + 262144);               // 512x512
    unsigned short* Wr0T = (unsigned short*)(wbase + 262144 + 524288);      // 6x128x256
    unsigned short* Wr1T = (unsigned short*)(wbase + 262144 + 524288 + 393216);            // 6x512x128
    unsigned short* Wh0T = (unsigned short*)(wbase + 262144 + 524288 + 393216 + 786432);   // 6x64x512
    unsigned short* ef   = e0;  // e1 fully consumed... e0 consumed by layer1 before route writes

    // rnn_states passthrough (first 1024*512 floats of d_out)
    hipMemcpyAsync(d_out, rnn, (size_t)1024 * 512 * sizeof(float),
                   hipMemcpyDeviceToDevice, stream);
    float* logits = (float*)d_out + (size_t)1024 * 512;

    // weight conversions
    transpose_bf16_kernel<<<dim3((DOBS * H + 255) / 256, 1), 256, 0, stream>>>(W0, W0T, DOBS, H);
    transpose_bf16_kernel<<<dim3((H * H + 255) / 256, 1), 256, 0, stream>>>(W1, W1T, H, H);
    transpose_bf16_kernel<<<dim3((DOBS * 128 + 255) / 256, NROLES), 256, 0, stream>>>(Wr0, Wr0T, DOBS, 128);
    transpose_bf16_kernel<<<dim3((128 * H + 255) / 256, NROLES), 256, 0, stream>>>(Wr1, Wr1T, 128, H);
    transpose_bf16_kernel<<<dim3((H * 64 + 255) / 256, NROLES), 256, 0, stream>>>(Wh0, Wh0T, H, 64);

    // pipeline
    ln_obs_kernel<<<TB / 8, 256, 0, stream>>>(obs, fn_scale, fn_bias, obs_n);
    gemm_relu_ln_512<<<TILES, 256, 0, stream>>>(obs_n, W0T, b0, ln0_s, ln0_b, e0, DOBS);
    gemm_relu_ln_512<<<TILES, 256, 0, stream>>>(e0, W1T, b1, ln1_s, ln1_b, e1, H);
    route_kernel<<<dim3(TILES, NROLES), 256, 0, stream>>>(obs_n, Wr0T, br0, Wr1T, br1,
                                                          e1, role_ids, ef);
    head_kernel<<<dim3(TILES, NROLES), 32, 0, stream>>>(ef, Wh0T, bh0, Wh1, bh1,
                                                        Wh2, bh2, role_ids, avail, logits);
}
